// MinkUNet_61366492725830
// MI455X (gfx1250) — matrix operand order repaired and checked by reading, not compiled, not run on hardware
//
#include <hip/hip_runtime.h>
#include <hip/hip_bf16.h>

// ---------------------------------------------------------------------------
// CDNA5 / gfx1250 WMMA types
// ---------------------------------------------------------------------------
typedef __attribute__((ext_vector_type(16))) __bf16 v16bf;
typedef __attribute__((ext_vector_type(8)))  float  v8f;

union V16U { uint4 q[2]; v16bf v; };
union V8FU { v8f v; float4 f[2]; };

// ---------------------------------------------------------------------------
// Weight prep: (co, ci, off) f32  ->  (co, off, ci) bf16.
// Reduction order everywhere is k = off*Ci + ci, so A rows are contiguous.
// ---------------------------------------------------------------------------
__global__ __launch_bounds__(256)
void wprep(const float* __restrict__ w, __bf16* __restrict__ o,
           int Ci, int K3, long long total)
{
    const long long i = (long long)blockIdx.x * 256 + threadIdx.x;
    if (i >= total) return;
    const int cik3 = Ci * K3;
    const int co   = (int)(i / cik3);
    const int r    = (int)(i - (long long)co * cik3);
    const int ci   = r / K3;
    const int off  = r - ci * K3;
    o[((size_t)co * K3 + off) * Ci + ci] = (__bf16)w[i];
}

// Input (1,3,64^3) f32 NCDHW -> bf16 NDHWC.
__global__ __launch_bounds__(256)
void x_to_bf16(const float* __restrict__ x, __bf16* __restrict__ o, int S)
{
    const int p = blockIdx.x * 256 + threadIdx.x;
    if (p >= S) return;
    o[(size_t)p * 3 + 0] = (__bf16)x[p];
    o[(size_t)p * 3 + 1] = (__bf16)x[(size_t)S + p];
    o[(size_t)p * 3 + 2] = (__bf16)x[(size_t)2 * S + p];
}

// ---------------------------------------------------------------------------
// Implicit-GEMM 3D conv on v_wmma_f32_16x16x32_bf16, bf16 NDHWC activations.
// One wave: 32 out-ch x 64 voxels = 2 A fragments x 4 B fragments -> 8 WMMAs
// per K-step, with all loads issued before the WMMA group so waitcnts stagger.
// For !RAGGED (Ci % 32 == 0) each 32-wide K-step lies inside one kernel
// offset, so A and B both load as 2x global_load_b128 of bf16.
// ---------------------------------------------------------------------------
template <int K, bool RAGGED>
__global__ __launch_bounds__(256)
void conv3d_wmma(const __bf16* __restrict__ act, const __bf16* __restrict__ wt,
                 float* __restrict__ out,
                 int Ci, int Co, int stride, int pad,
                 int Di, int Do_, int nTilesCo, int totalTiles)
{
    constexpr int K2 = K * K;
    constexpr int K3 = K2 * K;

    const int wave = blockIdx.x * (blockDim.x >> 5) + (threadIdx.x >> 5);
    if (wave >= totalTiles) return;

    const int lane   = threadIdx.x & 31;
    const int laneHi = lane >> 4;
    const int n      = lane & 15;

    const int coTile = wave % nTilesCo;     // 32 output channels per tile
    const int nTile  = wave / nTilesCo;

    // Four output voxels per lane.
    int bi[4], idB[4], ihB[4], iwB[4], p[4];
    bool inter = true;
#pragma unroll
    for (int j = 0; j < 4; ++j) {
        const int pj = nTile * 64 + j * 16 + n;
        p[j] = pj;
        const int od = pj / (Do_ * Do_);
        const int oh = (pj / Do_) % Do_;
        const int ow = pj % Do_;
        idB[j] = od * stride - pad;
        ihB[j] = oh * stride - pad;
        iwB[j] = ow * stride - pad;
        bi[j]  = (idB[j] * Di + ihB[j]) * Di + iwB[j];
        inter &= (idB[j] >= 0) & (ihB[j] >= 0) & (iwB[j] >= 0) &
                 (idB[j] + K <= Di) & (ihB[j] + K <= Di) & (iwB[j] + K <= Di);
    }

    v8f acc[2][4] = {{{}, {}, {}, {}}, {{}, {}, {}, {}}};

    if (!RAGGED) {
        const __bf16* __restrict__ wbase0 =
            wt + (size_t)(coTile * 32 + n) * K3 * Ci;
        const __bf16* __restrict__ wbase1 = wbase0 + (size_t)16 * K3 * Ci;
        const bool fast = (__builtin_amdgcn_ballot_w32(inter) == 0xffffffffu);

        if (fast) {
            for (int off = 0; off < K3; ++off) {
                const int kd = off / K2;
                const int r2 = off - kd * K2;
                const int kh = r2 / K;
                const int kw = r2 - kh * K;
                const int dpos = (kd * Di + kh) * Di + kw;
                int ip[4];
#pragma unroll
                for (int j = 0; j < 4; ++j) ip[j] = (bi[j] + dpos) * Ci;
                const __bf16* __restrict__ arow0 = wbase0 + (size_t)off * Ci;
                const __bf16* __restrict__ arow1 = wbase1 + (size_t)off * Ci;
                for (int cb = 0; cb < Ci; cb += 32) {
                    __builtin_prefetch(arow0 + cb + 32, 0, 1);
                    V16U a0, a1;
                    a0.q[0] = *(const uint4*)(arow0 + cb + 16 * laneHi);
                    a0.q[1] = *(const uint4*)(arow0 + cb + 8 + 16 * laneHi);
                    a1.q[0] = *(const uint4*)(arow1 + cb + 16 * laneHi);
                    a1.q[1] = *(const uint4*)(arow1 + cb + 8 + 16 * laneHi);
                    V16U b[4];
#pragma unroll
                    for (int j = 0; j < 4; ++j) {
                        const __bf16* bp = act + ip[j] + cb + 16 * laneHi;
                        b[j].q[0] = *(const uint4*)bp;
                        b[j].q[1] = *(const uint4*)(bp + 8);
                    }
#pragma unroll
                    for (int j = 0; j < 4; ++j) {
                        acc[0][j] = __builtin_amdgcn_wmma_f32_16x16x32_bf16(
                            false, a0.v, false, b[j].v, (short)0, acc[0][j],
                            false, false);
                        acc[1][j] = __builtin_amdgcn_wmma_f32_16x16x32_bf16(
                            false, a1.v, false, b[j].v, (short)0, acc[1][j],
                            false, false);
                    }
                }
            }
        } else {
            const uint4 z4 = {0u, 0u, 0u, 0u};
            for (int off = 0; off < K3; ++off) {
                const int kd = off / K2;
                const int r2 = off - kd * K2;
                const int kh = r2 / K;
                const int kw = r2 - kh * K;
                int ip[4];
                bool ok[4];
#pragma unroll
                for (int j = 0; j < 4; ++j) {
                    const int id = idB[j] + kd;
                    const int ih = ihB[j] + kh;
                    const int iw = iwB[j] + kw;
                    ok[j] = ((unsigned)id < (unsigned)Di) &
                            ((unsigned)ih < (unsigned)Di) &
                            ((unsigned)iw < (unsigned)Di);
                    ip[j] = ok[j] ? ((id * Di + ih) * Di + iw) * Ci : 0;
                }
                const __bf16* __restrict__ arow0 = wbase0 + (size_t)off * Ci;
                const __bf16* __restrict__ arow1 = wbase1 + (size_t)off * Ci;
                for (int cb = 0; cb < Ci; cb += 32) {
                    V16U a0, a1;
                    a0.q[0] = *(const uint4*)(arow0 + cb + 16 * laneHi);
                    a0.q[1] = *(const uint4*)(arow0 + cb + 8 + 16 * laneHi);
                    a1.q[0] = *(const uint4*)(arow1 + cb + 16 * laneHi);
                    a1.q[1] = *(const uint4*)(arow1 + cb + 8 + 16 * laneHi);
                    V16U b[4];
#pragma unroll
                    for (int j = 0; j < 4; ++j) {
                        const __bf16* bp = act + ip[j] + cb + 16 * laneHi;
                        b[j].q[0] = *(const uint4*)bp;
                        b[j].q[1] = *(const uint4*)(bp + 8);
                        if (!ok[j]) { b[j].q[0] = z4; b[j].q[1] = z4; }
                    }
#pragma unroll
                    for (int j = 0; j < 4; ++j) {
                        acc[0][j] = __builtin_amdgcn_wmma_f32_16x16x32_bf16(
                            false, a0.v, false, b[j].v, (short)0, acc[0][j],
                            false, false);
                        acc[1][j] = __builtin_amdgcn_wmma_f32_16x16x32_bf16(
                            false, a1.v, false, b[j].v, (short)0, acc[1][j],
                            false, false);
                    }
                }
            }
        }
    } else {
        // ---- ragged stem path (Ci==3): element-wise guarded gather ----
        const int CiK3 = Ci * K3;
        const __bf16* __restrict__ wrow0 =
            wt + (size_t)(coTile * 32 + n) * CiK3;
        const __bf16* __restrict__ wrow1 = wrow0 + (size_t)16 * CiK3;
        const int steps = (CiK3 + 31) >> 5;
        for (int s = 0; s < steps; ++s) {
            const int kBase = s << 5;
            v16bf a0, a1;
#pragma unroll
            for (int e = 0; e < 16; ++e) {
                const int ka = kBase + e + laneHi * 16;
                const bool oka = ka < CiK3;
                const __bf16 av0 = wrow0[oka ? ka : 0];
                const __bf16 av1 = wrow1[oka ? ka : 0];
                a0[e] = oka ? av0 : (__bf16)0.f;
                a1[e] = oka ? av1 : (__bf16)0.f;
            }
            int kdE[16], khE[16], kwE[16], ciE[16];
            bool okkE[16];
#pragma unroll
            for (int e = 0; e < 16; ++e) {
                const int kb  = kBase + e + laneHi * 16;
                const bool ok = kb < CiK3;
                const int kbc = ok ? kb : 0;
                const int off = kbc / Ci;
                ciE[e]  = kbc - off * Ci;
                kdE[e]  = off / K2;
                const int r2 = off - kdE[e] * K2;
                khE[e]  = r2 / K;
                kwE[e]  = r2 - khE[e] * K;
                okkE[e] = ok;
            }
#pragma unroll
            for (int j = 0; j < 4; ++j) {
                v16bf b;
#pragma unroll
                for (int e = 0; e < 16; ++e) {
                    const int id = idB[j] + kdE[e];
                    const int ih = ihB[j] + khE[e];
                    const int iw = iwB[j] + kwE[e];
                    const bool ok = okkE[e] &
                                    ((unsigned)id < (unsigned)Di) &
                                    ((unsigned)ih < (unsigned)Di) &
                                    ((unsigned)iw < (unsigned)Di);
                    const int addr = ((id * Di + ih) * Di + iw) * Ci + ciE[e];
                    const __bf16 bv = act[ok ? addr : 0];
                    b[e] = ok ? bv : (__bf16)0.f;
                }
                acc[0][j] = __builtin_amdgcn_wmma_f32_16x16x32_bf16(
                    false, a0, false, b, (short)0, acc[0][j], false, false);
                acc[1][j] = __builtin_amdgcn_wmma_f32_16x16x32_bf16(
                    false, a1, false, b, (short)0, acc[1][j], false, false);
            }
        }
    }

    // D store: 8 consecutive channels per lane per tile half -> 2x float4.
#pragma unroll
    for (int t = 0; t < 2; ++t)
#pragma unroll
        for (int j = 0; j < 4; ++j) {
            V8FU u; u.v = acc[t][j];
            float* op = out + (size_t)p[j] * Co + coTile * 32 + t * 16 +
                        laneHi * 8;
            *(float4*)op       = u.f[0];
            *((float4*)op + 1) = u.f[1];
        }
}

// ---------------------------------------------------------------------------
// 2x2x2 stride-2 transposed conv (kernel==stride): GEMM over Ci at a fixed
// (kd,kh,kw) per wave; 32 out-ch x 64 voxels per wave; scatter to 2x grid.
// ---------------------------------------------------------------------------
__global__ __launch_bounds__(256)
void deconv3d_wmma(const __bf16* __restrict__ act, const __bf16* __restrict__ wt,
                   float* __restrict__ out,
                   int Ci, int Co, int Di, int nTilesCo, int totalTiles)
{
    const int wave = blockIdx.x * (blockDim.x >> 5) + (threadIdx.x >> 5);
    if (wave >= totalTiles) return;

    const int lane   = threadIdx.x & 31;
    const int laneHi = lane >> 4;
    const int n      = lane & 15;

    const int kidx   = wave & 7;
    const int rest   = wave >> 3;
    const int coTile = rest % nTilesCo;
    const int iTile  = rest / nTilesCo;

    const int kd = kidx >> 2, kh = (kidx >> 1) & 1, kw = kidx & 1;
    const int Do_ = Di * 2;

    int p[4];
#pragma unroll
    for (int j = 0; j < 4; ++j) p[j] = iTile * 64 + j * 16 + n;

    const __bf16* __restrict__ arow0 =
        wt + ((size_t)(coTile * 32 + n) * 8 + kidx) * Ci;
    const __bf16* __restrict__ arow1 = arow0 + (size_t)16 * 8 * Ci;

    v8f acc[2][4] = {{{}, {}, {}, {}}, {{}, {}, {}, {}}};
    for (int cb = 0; cb < Ci; cb += 32) {
        V16U a0, a1;
        a0.q[0] = *(const uint4*)(arow0 + cb + 16 * laneHi);
        a0.q[1] = *(const uint4*)(arow0 + cb + 8 + 16 * laneHi);
        a1.q[0] = *(const uint4*)(arow1 + cb + 16 * laneHi);
        a1.q[1] = *(const uint4*)(arow1 + cb + 8 + 16 * laneHi);
        V16U b[4];
#pragma unroll
        for (int j = 0; j < 4; ++j) {
            const __bf16* bp = act + (size_t)p[j] * Ci + cb + 16 * laneHi;
            b[j].q[0] = *(const uint4*)bp;
            b[j].q[1] = *(const uint4*)(bp + 8);
        }
#pragma unroll
        for (int j = 0; j < 4; ++j) {
            acc[0][j] = __builtin_amdgcn_wmma_f32_16x16x32_bf16(
                false, a0.v, false, b[j].v, (short)0, acc[0][j], false, false);
            acc[1][j] = __builtin_amdgcn_wmma_f32_16x16x32_bf16(
                false, a1.v, false, b[j].v, (short)0, acc[1][j], false, false);
        }
    }

#pragma unroll
    for (int j = 0; j < 4; ++j) {
        const int id = p[j] / (Di * Di);
        const int ih = (p[j] / Di) % Di;
        const int iw = p[j] % Di;
        const size_t op =
            ((size_t)(2 * id + kd) * Do_ + (2 * ih + kh)) * Do_ + (2 * iw + kw);
#pragma unroll
        for (int t = 0; t < 2; ++t) {
            V8FU u; u.v = acc[t][j];
            float* o = out + op * Co + coTile * 32 + t * 16 + laneHi * 8;
            *(float4*)o       = u.f[0];
            *((float4*)o + 1) = u.f[1];
        }
    }
}

// ---------------------------------------------------------------------------
// BatchNorm over NDHWC f32: LDS per-channel accumulation (ds_add_f32) +
// global atomics, then finalize, then fused apply (+resid)(+ReLU) -> bf16.
// ---------------------------------------------------------------------------
__global__ __launch_bounds__(256)
void zero_k(float* __restrict__ p, int nelem)
{
    const int i = blockIdx.x * 256 + threadIdx.x;
    if (i < nelem) p[i] = 0.f;
}

__global__ __launch_bounds__(256)
void bn_stats(const float* __restrict__ x, float* __restrict__ gsum,
              float* __restrict__ gsq, int C, int quads, long long total4)
{
    __shared__ float sS[384];
    __shared__ float sQ[384];
    for (int t = threadIdx.x; t < C; t += blockDim.x) { sS[t] = 0.f; sQ[t] = 0.f; }
    __syncthreads();
    const long long stride = (long long)gridDim.x * blockDim.x;
    for (long long i = (long long)blockIdx.x * blockDim.x + threadIdx.x;
         i < total4; i += stride) {
        const float4 v = ((const float4*)x)[i];
        const int c4 = (int)(i % quads) * 4;
        atomicAdd(&sS[c4 + 0], v.x); atomicAdd(&sQ[c4 + 0], v.x * v.x);
        atomicAdd(&sS[c4 + 1], v.y); atomicAdd(&sQ[c4 + 1], v.y * v.y);
        atomicAdd(&sS[c4 + 2], v.z); atomicAdd(&sQ[c4 + 2], v.z * v.z);
        atomicAdd(&sS[c4 + 3], v.w); atomicAdd(&sQ[c4 + 3], v.w * v.w);
    }
    __syncthreads();
    for (int t = threadIdx.x; t < C; t += blockDim.x) {
        atomicAdd(&gsum[t], sS[t]);
        atomicAdd(&gsq[t],  sQ[t]);
    }
}

__global__ __launch_bounds__(256)
void bn_finalize(const float* __restrict__ gsum, const float* __restrict__ gsq,
                 const float* __restrict__ g, const float* __restrict__ b,
                 float* __restrict__ scale, float* __restrict__ shift,
                 int C, float invS)
{
    const int c = blockIdx.x * 256 + threadIdx.x;
    if (c >= C) return;
    const float mean = gsum[c] * invS;
    const float var  = gsq[c] * invS - mean * mean;
    const float sc   = g[c] * rsqrtf(var + 1e-5f);
    scale[c] = sc;
    shift[c] = b[c] - mean * sc;
}

__global__ __launch_bounds__(256)
void bn_apply(const float* __restrict__ x, const float* __restrict__ scale,
              const float* __restrict__ shift, const __bf16* __restrict__ resid,
              __bf16* __restrict__ outb, int outStride,
              float* __restrict__ out_t,
              int S, int C, int quads, long long total4, int relu)
{
    const long long i = (long long)blockIdx.x * blockDim.x + threadIdx.x;
    if (i >= total4) return;
    const long long pl = i / quads;
    const int p  = (int)pl;
    const int c4 = (int)(i - pl * quads) * 4;

    float4 v = ((const float4*)x)[i];
    const float4 sc = *(const float4*)(scale + c4);
    const float4 sf = *(const float4*)(shift + c4);
    v.x = v.x * sc.x + sf.x; v.y = v.y * sc.y + sf.y;
    v.z = v.z * sc.z + sf.z; v.w = v.w * sc.w + sf.w;
    if (resid) {
        union { uint2 u; __bf16 h[4]; } r;
        r.u = *(const uint2*)(resid + (size_t)p * C + c4);
        v.x += (float)r.h[0]; v.y += (float)r.h[1];
        v.z += (float)r.h[2]; v.w += (float)r.h[3];
    }
    if (relu) {
        v.x = fmaxf(v.x, 0.f); v.y = fmaxf(v.y, 0.f);
        v.z = fmaxf(v.z, 0.f); v.w = fmaxf(v.w, 0.f);
    }
    union { uint2 u; __bf16 h[4]; } o;
    o.h[0] = (__bf16)v.x; o.h[1] = (__bf16)v.y;
    o.h[2] = (__bf16)v.z; o.h[3] = (__bf16)v.w;
    *(uint2*)(outb + (size_t)p * outStride + c4) = o.u;
    if (out_t) {   // final layer: transpose back to NCDHW f32
        out_t[(size_t)(c4 + 0) * S + p] = v.x;
        out_t[(size_t)(c4 + 1) * S + p] = v.y;
        out_t[(size_t)(c4 + 2) * S + p] = v.z;
        out_t[(size_t)(c4 + 3) * S + p] = v.w;
    }
}

// Copy skip channels into the tail of a concat buffer (bf16 NDHWC).
__global__ __launch_bounds__(256)
void cat_copy(const __bf16* __restrict__ skip, __bf16* __restrict__ cat,
              int C2, int Ctot, int C1, long long total4)
{
    const long long i = (long long)blockIdx.x * blockDim.x + threadIdx.x;
    if (i >= total4) return;
    const int q  = C2 >> 2;
    const long long pl = i / q;
    const int c4 = (int)(i - pl * q) * 4;
    *(uint2*)(cat + (size_t)pl * Ctot + C1 + c4) =
        *(const uint2*)(skip + (size_t)pl * C2 + c4);
}

// ---------------------------------------------------------------------------
// Host-side helpers
// ---------------------------------------------------------------------------
struct CP { const float *w, *g, *b; };
struct RP { const float *w1, *g1, *b1, *w2, *g2, *b2, *wd, *gd, *bd; };
struct RB { const __bf16 *w1, *w2, *wd; };  // prepped bf16 weights

static void conv(hipStream_t st, const __bf16* in, const __bf16* wt, float* out,
                 int Ci, int Co, int K, int stride, int pad, int Di, int Do_)
{
    const int nCo    = Co / 32;
    const int S      = Do_ * Do_ * Do_;
    const int total  = nCo * (S / 64);
    const int blocks = (total + 7) / 8;
    if (K == 3 && Ci == 3)
        conv3d_wmma<3, true><<<blocks, 256, 0, st>>>(in, wt, out, Ci, Co, stride,
                                                     pad, Di, Do_, nCo, total);
    else if (K == 3)
        conv3d_wmma<3, false><<<blocks, 256, 0, st>>>(in, wt, out, Ci, Co, stride,
                                                      pad, Di, Do_, nCo, total);
    else if (K == 2)
        conv3d_wmma<2, false><<<blocks, 256, 0, st>>>(in, wt, out, Ci, Co, stride,
                                                      pad, Di, Do_, nCo, total);
    else
        conv3d_wmma<1, false><<<blocks, 256, 0, st>>>(in, wt, out, Ci, Co, stride,
                                                      pad, Di, Do_, nCo, total);
}

static void deconv(hipStream_t st, const __bf16* in, const __bf16* wt,
                   float* out, int Ci, int Co, int Di)
{
    const int nCo   = Co / 32;
    const int inS   = Di * Di * Di;
    const int total = nCo * (inS / 64) * 8;
    deconv3d_wmma<<<(total + 7) / 8, 256, 0, st>>>(in, wt, out, Ci, Co, Di, nCo,
                                                   total);
}

struct BNBuf { float *gsum, *gsq, *scale, *shift; };

static void bn(hipStream_t st, const float* x, const float* g, const float* b,
               const BNBuf& bb, int C, int S, const __bf16* resid, int relu,
               __bf16* outb, int outStride, float* out_t)
{
    zero_k<<<4, 256, 0, st>>>(bb.gsum, 1024);       // gsum(512)+gsq(512)
    const int quads = C >> 2;
    const long long total4 = ((long long)S * C) >> 2;
    int blocks = (int)((total4 + 255) / 256);
    if (blocks > 2048) blocks = 2048;
    bn_stats<<<blocks, 256, 0, st>>>(x, bb.gsum, bb.gsq, C, quads, total4);
    bn_finalize<<<2, 256, 0, st>>>(bb.gsum, bb.gsq, g, b, bb.scale, bb.shift, C,
                                   1.f / (float)S);
    bn_apply<<<(int)((total4 + 255) / 256), 256, 0, st>>>(
        x, bb.scale, bb.shift, resid, outb, outStride, out_t, S, C, quads,
        total4, relu);
}

static void cbr(hipStream_t st, const __bf16* in, const __bf16* wp,
                const CP& p, int Ci, int Co, int K, int stride, int pad,
                int Di, int Do_, float* F, __bf16* outAct, const BNBuf& bb)
{
    conv(st, in, wp, F, Ci, Co, K, stride, pad, Di, Do_);
    bn(st, F, p.g, p.b, bb, Co, Do_ * Do_ * Do_, nullptr, 1, outAct, Co, nullptr);
}

static void resblock(hipStream_t st, const __bf16* in, const RP& r, const RB& w,
                     int Ci, int Co, int D, __bf16* outAct, __bf16* t1Act,
                     __bf16* scAct, float* F0, float* F1, const BNBuf& bb,
                     float* out_t)
{
    const int S = D * D * D;
    conv(st, in, w.w1, F0, Ci, Co, 3, 1, 1, D, D);
    bn(st, F0, r.g1, r.b1, bb, Co, S, nullptr, 1, t1Act, Co, nullptr);
    conv(st, t1Act, w.w2, F1, Co, Co, 3, 1, 1, D, D);
    const __bf16* resid;
    if (r.wd) {                                  // 1x1x1 projection shortcut
        conv(st, in, w.wd, F0, Ci, Co, 1, 1, 0, D, D);
        bn(st, F0, r.gd, r.bd, bb, Co, S, nullptr, 0, scAct, Co, nullptr);
        resid = scAct;
    } else {
        resid = in;
    }
    bn(st, F1, r.g2, r.b2, bb, Co, S, resid, 1, outAct, Co, out_t);
}

// ---------------------------------------------------------------------------
// kernel_launch: full MinkUNet forward pass.
// ---------------------------------------------------------------------------
extern "C" void kernel_launch(void* const* d_in, const int* in_sizes, int n_in,
                              void* d_out, int out_size, void* d_ws,
                              size_t ws_size, hipStream_t stream)
{
    (void)in_sizes; (void)n_in; (void)out_size; (void)ws_size;
    hipStream_t st = stream;

    int ii = 0;
    auto N = [&]() { return (const float*)d_in[ii++]; };

    const float* x = N();                                   // (1,3,64,64,64)
    CP stem1{N(), N(), N()};
    CP stem2{N(), N(), N()};
    auto RD = [&](bool proj) {
        RP r;
        r.w1 = N(); r.g1 = N(); r.b1 = N();
        r.w2 = N(); r.g2 = N(); r.b2 = N();
        if (proj) { r.wd = N(); r.gd = N(); r.bd = N(); }
        else      { r.wd = nullptr; r.gd = nullptr; r.bd = nullptr; }
        return r;
    };
    CP d1{N(), N(), N()};   RP s1r1 = RD(false), s1r2 = RD(false);
    CP d2{N(), N(), N()};   RP s2r1 = RD(true),  s2r2 = RD(false);
    CP d3{N(), N(), N()};   RP s3r1 = RD(true),  s3r2 = RD(false);
    CP d4{N(), N(), N()};   RP s4r1 = RD(true),  s4r2 = RD(false);
    CP u1de{N(), N(), N()}; RP u1r1 = RD(true),  u1r2 = RD(false);
    CP u2de{N(), N(), N()}; RP u2r1 = RD(true),  u2r2 = RD(false);
    CP u3de{N(), N(), N()}; RP u3r1 = RD(true),  u3r2 = RD(false);
    CP u4de{N(), N(), N()}; RP u4r1 = RD(true),  u4r2 = RD(false);

    // ---- workspace carve ----
    char* pb = (char*)d_ws;
    auto carve = [&](size_t bytes) {
        void* p = pb;
        pb += (bytes + 255) & ~(size_t)255;
        return p;
    };
    const size_t MAXE = 33554432;                    // 128ch * 64^3
    __bf16* A0  = (__bf16*)carve(MAXE * 2);
    __bf16* A1  = (__bf16*)carve(MAXE * 2);
    __bf16* A2  = (__bf16*)carve(MAXE * 2);
    __bf16* A3  = (__bf16*)carve(MAXE * 2);
    float*  F0  = (float*)carve((size_t)25165824 * 4);
    float*  F1  = (float*)carve((size_t)25165824 * 4);
    __bf16* X0b = (__bf16*)carve((size_t)32 * 262144 * 2);
    __bf16* X1b = (__bf16*)carve((size_t)32 * 32768 * 2);
    __bf16* X2b = (__bf16*)carve((size_t)64 * 4096 * 2);
    __bf16* X3b = (__bf16*)carve((size_t)128 * 512 * 2);
    __bf16* Xin = (__bf16*)carve((size_t)3 * 262144 * 2);
    BNBuf bb;
    bb.gsum  = (float*)carve(1024 * 4);              // gsum(512)+gsq(512)
    bb.gsq   = bb.gsum + 512;
    bb.scale = (float*)carve(512 * 4);
    bb.shift = (float*)carve(512 * 4);
    __bf16* wpool = (__bf16*)carve((size_t)24 * 1024 * 1024 * 2);

    // ---- weight prep (f32 (co,ci,off) -> bf16 (co,off,ci)) ----
    size_t wo = 0;
    auto prep = [&](const float* w, int Co_, int Ci_, int K3_) {
        __bf16* dst = wpool + wo;
        const long long total = (long long)Co_ * Ci_ * K3_;
        wo += (size_t)((total + 7) & ~7ll);
        wprep<<<(int)((total + 255) / 256), 256, 0, st>>>(w, dst, Ci_, K3_, total);
        return (const __bf16*)dst;
    };
    auto prepR = [&](const RP& r, int Ci_, int Co_) {
        RB o;
        o.w1 = prep(r.w1, Co_, Ci_, 27);
        o.w2 = prep(r.w2, Co_, Co_, 27);
        o.wd = r.wd ? prep(r.wd, Co_, Ci_, 1) : nullptr;
        return o;
    };
    const __bf16* stem1w = prep(stem1.w, 32, 3, 27);
    const __bf16* stem2w = prep(stem2.w, 32, 32, 27);
    const __bf16* d1w = prep(d1.w, 32, 32, 8);   RB s1r1w = prepR(s1r1, 32, 32),  s1r2w = prepR(s1r2, 32, 32);
    const __bf16* d2w = prep(d2.w, 32, 32, 8);   RB s2r1w = prepR(s2r1, 32, 64),  s2r2w = prepR(s2r2, 64, 64);
    const __bf16* d3w = prep(d3.w, 64, 64, 8);   RB s3r1w = prepR(s3r1, 64, 128), s3r2w = prepR(s3r2, 128, 128);
    const __bf16* d4w = prep(d4.w, 128, 128, 8); RB s4r1w = prepR(s4r1, 128, 256), s4r2w = prepR(s4r2, 256, 256);
    const __bf16* u1w = prep(u1de.w, 256, 256, 8); RB u1r1w = prepR(u1r1, 384, 256), u1r2w = prepR(u1r2, 256, 256);
    const __bf16* u2w = prep(u2de.w, 128, 256, 8); RB u2r1w = prepR(u2r1, 192, 128), u2r2w = prepR(u2r2, 128, 128);
    const __bf16* u3w = prep(u3de.w, 96, 128, 8);  RB u3r1w = prepR(u3r1, 128, 96),  u3r2w = prepR(u3r2, 96, 96);
    const __bf16* u4w = prep(u4de.w, 96, 96, 8);   RB u4r1w = prepR(u4r1, 128, 96),  u4r2w = prepR(u4r2, 96, 96);

    // ---- stem ----
    x_to_bf16<<<(262144 + 255) / 256, 256, 0, st>>>(x, Xin, 262144);
    cbr(st, Xin, stem1w, stem1, 3, 32, 3, 1, 1, 64, 64, F0, A0, bb);
    cbr(st, A0, stem2w, stem2, 32, 32, 3, 1, 1, 64, 64, F0, X0b, bb);   // x0

    // ---- encoder ----
    cbr(st, X0b, d1w, d1, 32, 32, 2, 2, 0, 64, 32, F0, A0, bb);
    resblock(st, A0, s1r1, s1r1w, 32, 32, 32, A2, A1, A3, F0, F1, bb, nullptr);
    resblock(st, A2, s1r2, s1r2w, 32, 32, 32, X1b, A1, A3, F0, F1, bb, nullptr);

    cbr(st, X1b, d2w, d2, 32, 32, 2, 2, 0, 32, 16, F0, A0, bb);
    resblock(st, A0, s2r1, s2r1w, 32, 64, 16, A2, A1, A3, F0, F1, bb, nullptr);
    resblock(st, A2, s2r2, s2r2w, 64, 64, 16, X2b, A1, A3, F0, F1, bb, nullptr);

    cbr(st, X2b, d3w, d3, 64, 64, 2, 2, 0, 16, 8, F0, A0, bb);
    resblock(st, A0, s3r1, s3r1w, 64, 128, 8, A2, A1, A3, F0, F1, bb, nullptr);
    resblock(st, A2, s3r2, s3r2w, 128, 128, 8, X3b, A1, A3, F0, F1, bb, nullptr);

    cbr(st, X3b, d4w, d4, 128, 128, 2, 2, 0, 8, 4, F0, A0, bb);
    resblock(st, A0, s4r1, s4r1w, 128, 256, 4, A2, A1, A3, F0, F1, bb, nullptr);
    resblock(st, A2, s4r2, s4r2w, 256, 256, 4, A0, A1, A3, F0, F1, bb, nullptr);

    // ---- up1: 256@4 -> 256@8, skip x3 = 128@8 ; cat = A1 (384ch) ----
    deconv(st, A0, u1w, F0, 256, 256, 4);
    bn(st, F0, u1de.g, u1de.b, bb, 256, 512, nullptr, 1, A1, 384, nullptr);
    cat_copy<<<(int)(((long long)512 * 128 / 4 + 255) / 256), 256, 0, st>>>(
        X3b, A1, 128, 384, 256, (long long)512 * 128 / 4);
    resblock(st, A1, u1r1, u1r1w, 384, 256, 8, A2, A3, A0, F0, F1, bb, nullptr);
    resblock(st, A2, u1r2, u1r2w, 256, 256, 8, A1, A3, A0, F0, F1, bb, nullptr);

    // ---- up2: 256@8 -> 128@16, skip x2 = 64@16 ; cat = A0 (192ch) ----
    deconv(st, A1, u2w, F0, 256, 128, 8);
    bn(st, F0, u2de.g, u2de.b, bb, 128, 4096, nullptr, 1, A0, 192, nullptr);
    cat_copy<<<(int)(((long long)4096 * 64 / 4 + 255) / 256), 256, 0, st>>>(
        X2b, A0, 64, 192, 128, (long long)4096 * 64 / 4);
    resblock(st, A0, u2r1, u2r1w, 192, 128, 16, A2, A3, A1, F0, F1, bb, nullptr);
    resblock(st, A2, u2r2, u2r2w, 128, 128, 16, A0, A3, A1, F0, F1, bb, nullptr);

    // ---- up3: 128@16 -> 96@32, skip x1 = 32@32 ; cat = A1 (128ch) ----
    deconv(st, A0, u3w, F0, 128, 96, 16);
    bn(st, F0, u3de.g, u3de.b, bb, 96, 32768, nullptr, 1, A1, 128, nullptr);
    cat_copy<<<(int)(((long long)32768 * 32 / 4 + 255) / 256), 256, 0, st>>>(
        X1b, A1, 32, 128, 96, (long long)32768 * 32 / 4);
    resblock(st, A1, u3r1, u3r1w, 128, 96, 32, A2, A3, A0, F0, F1, bb, nullptr);
    resblock(st, A2, u3r2, u3r2w, 96, 96, 32, A1, A3, A0, F0, F1, bb, nullptr);

    // ---- up4: 96@32 -> 96@64, skip x0 = 32@64 ; cat = A0 (128ch) ----
    deconv(st, A1, u4w, F0, 96, 96, 32);
    bn(st, F0, u4de.g, u4de.b, bb, 96, 262144, nullptr, 1, A0, 128, nullptr);
    cat_copy<<<(int)(((long long)262144 * 32 / 4 + 255) / 256), 256, 0, st>>>(
        X0b, A0, 32, 128, 96, (long long)262144 * 32 / 4);
    resblock(st, A0, u4r1, u4r1w, 128, 96, 64, A2, A3, A1, F0, F1, bb, nullptr);
    // final: BN-apply also writes f32 NCDHW into d_out
    resblock(st, A2, u4r2, u4r2w, 96, 96, 64, A1, A3, A0, F0, F1, bb,
             (float*)d_out);
}
